// RotatTransformerEncoder_58643483460199
// MI455X (gfx1250) — compile-verified, matrix operand order checked by reading, not run on hardware
//
#include <hip/hip_runtime.h>
#include <cstdint>

// ---------------------------------------------------------------------------
// Problem constants (from the reference)
// ---------------------------------------------------------------------------
#define D_    128
#define H_    8
#define S_    127          // 2*SHOT+1
#define E_    1024         // D*H
#define FFN_  4096
#define NB_   256          // batch
#define M_    (NB_ * S_)   // 32512 rows, divisible by 128
#define P_    65           // SHOT+2
#define QK_SCALE 0.0625f   // (D*SCALE)^-0.5 = 256^-0.5

#define CDIV(a, b) (((a) + (b) - 1) / (b))

typedef __attribute__((ext_vector_type(4)))  unsigned int uivec4;
typedef __attribute__((ext_vector_type(16))) __bf16       v16bf;
typedef __attribute__((ext_vector_type(8)))  float        v8f;

union FragBF { uivec4 u[2]; v16bf v; };   // 32 bytes, trivially copyable

// ---------------------------------------------------------------------------
// bf16 helpers (buffers held as unsigned short)
// ---------------------------------------------------------------------------
__device__ __forceinline__ float bf2f(unsigned short h) {
  unsigned u = ((unsigned)h) << 16;
  return __builtin_bit_cast(float, u);
}
__device__ __forceinline__ unsigned short f2bf(float f) {
  unsigned u = __builtin_bit_cast(unsigned, f);
  u += 0x7FFFu + ((u >> 16) & 1u);      // round-to-nearest-even
  return (unsigned short)(u >> 16);
}

__device__ __forceinline__ float wave_sum(float v) {
#pragma unroll
  for (int m = 16; m >= 1; m >>= 1) v += __shfl_xor(v, m, 32);
  return v;
}
__device__ __forceinline__ float wave_max(float v) {
#pragma unroll
  for (int m = 16; m >= 1; m >>= 1) v = fmaxf(v, __shfl_xor(v, m, 32));
  return v;
}

// ---------------------------------------------------------------------------
// fp32 -> bf16 conversion (weights / generic)
// ---------------------------------------------------------------------------
__global__ void convert_bf16_kernel(const float* __restrict__ s,
                                    unsigned short* __restrict__ d, size_t n) {
  size_t i = (size_t)blockIdx.x * 256 + threadIdx.x;
  if (i < n) d[i] = f2bf(s[i]);
}

// ---------------------------------------------------------------------------
// x init: x[n,s,:] = repeat_H(s==0 ? cls : entity[n,s-1]); writes f32 + bf16
// ---------------------------------------------------------------------------
__global__ void init_x_kernel(const float* __restrict__ entity,
                              const float* __restrict__ cls,
                              float* __restrict__ x,
                              unsigned short* __restrict__ xb) {
  size_t idx = (size_t)blockIdx.x * 256 + threadIdx.x;
  if (idx >= (size_t)M_ * E_) return;
  size_t m = idx >> 10;            // / E_
  int e = (int)(idx & 1023);
  int s = (int)(m % S_);
  int n = (int)(m / S_);
  int d = e & (D_ - 1);
  float v = (s == 0) ? cls[d]
                     : entity[((size_t)n * (S_ - 1) + (s - 1)) * D_ + d];
  x[idx]  = v;
  xb[idx] = f2bf(v);
}

// ---------------------------------------------------------------------------
// pos-bias pipeline (tiny precompute)
// ---------------------------------------------------------------------------
__global__ __launch_bounds__(128)
void ln_pos_kernel(const float* __restrict__ ppe, const float* __restrict__ g,
                   const float* __restrict__ b, float* __restrict__ wgt128) {
  int row = blockIdx.x, tid = threadIdx.x;
  float v = ppe[row * 128 + tid];
  __shared__ float red[8];
  float s  = wave_sum(v);
  float s2 = wave_sum(v * v);
  if ((tid & 31) == 0) { red[tid >> 5] = s; red[4 + (tid >> 5)] = s2; }
  __syncthreads();
  float ts  = red[0] + red[1] + red[2] + red[3];
  float ts2 = red[4] + red[5] + red[6] + red[7];
  float mean = ts * (1.0f / 128.0f);
  float var  = ts2 * (1.0f / 128.0f) - mean * mean;
  float inv  = rsqrtf(var + 1e-5f);
  wgt128[row * 128 + tid] = (v - mean) * inv * g[tid] + b[tid];
}

// pq[h,p,d] = (tile_H(wgt128) @ W^T + bias)[p, h*128+d] * scale
__global__ __launch_bounds__(128)
void pos_lin_kernel(const float* __restrict__ wgt128, const float* __restrict__ W,
                    const float* __restrict__ bvec, float* __restrict__ out,
                    float scale) {
  int p = blockIdx.x;       // 0..64
  int h = blockIdx.y;       // 0..7
  int d = threadIdx.x;      // 0..127
  int e = h * 128 + d;
  const float* wrow = &W[(size_t)e * E_];
  const float* grow = &wgt128[p * 128];
  float acc = 0.f;
  for (int k = 0; k < 128; ++k) {
    float wsum = 0.f;
#pragma unroll
    for (int r = 0; r < H_; ++r) wsum += wrow[r * 128 + k];
    acc += grow[k] * wsum;
  }
  out[((size_t)h * P_ + p) * 128 + d] = (acc + bvec[e]) * scale;
}

// bias[h,s,t] per the reference's repeat/slice/override construction
__global__ __launch_bounds__(128)
void bias_build_kernel(const float* __restrict__ pq, const float* __restrict__ pk,
                       float* __restrict__ bias) {
  int h = blockIdx.x, s = blockIdx.y;
  for (int t = threadIdx.x; t < S_; t += 128) {
    int ps, pt;
    if (s == 0)      { ps = 0; pt = 0; }            // cls2o
    else if (t == 0) { ps = 1; pt = 1; }            // o2cls
    else             { ps = 1 + ((s + 1) >> 1); pt = 1 + ((t + 1) >> 1); }
    const float* a = &pq[((size_t)h * P_ + ps) * 128];
    const float* c = &pk[((size_t)h * P_ + pt) * 128];
    float acc = 0.f;
    for (int d = 0; d < 128; ++d) acc += a[d] * c[d];
    bias[((size_t)h * S_ + s) * S_ + t] = acc;
  }
}

// ---------------------------------------------------------------------------
// WMMA GEMM: C[M,N] = A[M,K](bf16) @ B[N,K]^T(bf16) + bias[N]
// 256 threads (8 wave32), 128x128 tile, K-step 32.
// Ping-pong double-buffered LDS (one barrier per K-step, global loads overlap
// compute) + grouped B-fragment preload so WMMAs issue back-to-back.
// ---------------------------------------------------------------------------
#define BM 128
#define BN 128
#define BK 32
#define BKP 40   // padded LDS row stride (ushorts): 80B, 16B-aligned

template <bool RELU, bool OUTBF>
__global__ __launch_bounds__(256)
void gemm_bf16_kernel(const unsigned short* __restrict__ A,
                      const unsigned short* __restrict__ B,
                      const float* __restrict__ bias,
                      float* __restrict__ outF,
                      unsigned short* __restrict__ outB,
                      int M, int N, int K) {
  __shared__ unsigned short As[2][BM * BKP];
  __shared__ unsigned short Bs[2][BN * BKP];

  const int tid  = threadIdx.x;
  const int wid  = tid >> 5;         // wave 0..7 -> M sub-tile
  const int lane = tid & 31;
  const int hf   = lane >> 4;        // K-half selector (ISA 16-bit layout)
  const int r    = lane & 15;        // row (A) / col (B) within 16-tile
  const int bm   = blockIdx.y * BM;
  const int bn   = blockIdx.x * BN;

  const int ldr = tid >> 2;          // 0..63: tile row loaded by this thread
  const int ldc = (tid & 3) << 3;    // 0,8,16,24: element offset (16B chunks)

  const unsigned short* Ar0 = &A[(size_t)(bm + ldr)      * K + ldc];
  const unsigned short* Ar1 = &A[(size_t)(bm + ldr + 64) * K + ldc];
  const unsigned short* Br0 = &B[(size_t)(bn + ldr)      * K + ldc];
  const unsigned short* Br1 = &B[(size_t)(bn + ldr + 64) * K + ldc];

  // prologue: tile 0 -> LDS[0]
  uivec4 a0 = *(const uivec4*)&Ar0[0];
  uivec4 a1 = *(const uivec4*)&Ar1[0];
  uivec4 b0 = *(const uivec4*)&Br0[0];
  uivec4 b1 = *(const uivec4*)&Br1[0];
  *(uivec4*)&As[0][ldr * BKP + ldc]        = a0;
  *(uivec4*)&As[0][(ldr + 64) * BKP + ldc] = a1;
  *(uivec4*)&Bs[0][ldr * BKP + ldc]        = b0;
  *(uivec4*)&Bs[0][(ldr + 64) * BKP + ldc] = b1;
  __syncthreads();

  v8f acc[8] = {};
  const int nk = K / BK;
  int buf = 0;

  for (int t = 0; t < nk; ++t) {
    if (t + 1 < nk) {                // global loads for next tile (overlap)
      const int k0 = (t + 1) * BK;
      a0 = *(const uivec4*)&Ar0[k0];
      a1 = *(const uivec4*)&Ar1[k0];
      b0 = *(const uivec4*)&Br0[k0];
      b1 = *(const uivec4*)&Br1[k0];
    }

    FragBF af;
    const unsigned short* asrc = &As[buf][(16 * wid + r) * BKP + hf * 8];
    af.u[0] = *(const uivec4*)asrc;
    af.u[1] = *(const uivec4*)(asrc + 16);

    FragBF bfr[4];
#pragma unroll
    for (int j = 0; j < 4; ++j) {
      const unsigned short* bsrc = &Bs[buf][(16 * j + r) * BKP + hf * 8];
      bfr[j].u[0] = *(const uivec4*)bsrc;
      bfr[j].u[1] = *(const uivec4*)(bsrc + 16);
    }
#pragma unroll
    for (int j = 0; j < 4; ++j)
      acc[j] = __builtin_amdgcn_wmma_f32_16x16x32_bf16(
          false, af.v, false, bfr[j].v, (short)0, acc[j], false, false);
#pragma unroll
    for (int j = 0; j < 4; ++j) {
      const unsigned short* bsrc = &Bs[buf][(16 * (j + 4) + r) * BKP + hf * 8];
      bfr[j].u[0] = *(const uivec4*)bsrc;
      bfr[j].u[1] = *(const uivec4*)(bsrc + 16);
    }
#pragma unroll
    for (int j = 0; j < 4; ++j)
      acc[j + 4] = __builtin_amdgcn_wmma_f32_16x16x32_bf16(
          false, af.v, false, bfr[j].v, (short)0, acc[j + 4], false, false);

    if (t + 1 < nk) {                // store next tile into the other buffer
      *(uivec4*)&As[buf ^ 1][ldr * BKP + ldc]        = a0;
      *(uivec4*)&As[buf ^ 1][(ldr + 64) * BKP + ldc] = a1;
      *(uivec4*)&Bs[buf ^ 1][ldr * BKP + ldc]        = b0;
      *(uivec4*)&Bs[buf ^ 1][(ldr + 64) * BKP + ldc] = b1;
      buf ^= 1;
    }
    __syncthreads();
  }

  // Epilogue. C/D layout: lane -> N = r, VGPR i -> M = i + 8*hf.
#pragma unroll
  for (int j = 0; j < 8; ++j) {
    const int n = bn + 16 * j + r;
    const float bv = bias[n];
#pragma unroll
    for (int i = 0; i < 8; ++i) {
      const int m = bm + 16 * wid + 8 * hf + i;
      float v = acc[j][i] + bv;
      if (RELU) v = v > 0.f ? v : 0.f;
      if (OUTBF) outB[(size_t)m * N + n] = f2bf(v);
      else       outF[(size_t)m * N + n] = v;
    }
  }
}

// ---------------------------------------------------------------------------
// RoPE (in place on bf16 buffer), with optional q-scale folded in
// ---------------------------------------------------------------------------
__global__ void rope_kernel(unsigned short* __restrict__ t, float scale) {
  size_t idx = (size_t)blockIdx.x * 256 + threadIdx.x;   // pair index
  if (idx >= (size_t)M_ * E_ / 2) return;
  size_t m = idx >> 9;                 // 512 pairs per row
  int e = (int)(idx & 511) * 2;
  int s = (int)(m % S_);
  int i = (e & (D_ - 1)) >> 1;
  float p   = (s == 0) ? 0.f : (float)(1 + ((s - 1) & 1));
  float ang = p * __expf(-(float)i * 0.14391157f);       // ln(1e4)/64
  float c = __cosf(ang), sn = __sinf(ang);
  size_t base = m * E_ + e;
  float t0 = bf2f(t[base]), t1 = bf2f(t[base + 1]);
  t[base]     = f2bf((t0 * c - t1 * sn) * scale);
  t[base + 1] = f2bf((t1 * c + t0 * sn) * scale);
}

// ---------------------------------------------------------------------------
// Attention: one block per (n,h). WMMA scores + dual softmax + WMMA ctx.
// Dynamic LDS: sc[128*128] f32 (64KB) + vT[128*136] bf16 (34KB).
// q/k buffers are over-allocated by 16 rows so edge-tile fragment loads are
// in-bounds; garbage only reaches the unstored row/col of edge tiles.
// ---------------------------------------------------------------------------
#define VTP 136   // vT row stride (ushorts), 272B = 17*16
#define ATTN_LDS (128 * 128 * 4 + 128 * VTP * 2)

__device__ __forceinline__ bool is_masked(int s, int t) {
  if (s == 0) return false;           // row 0 fully unmasked
  return !((t > 0) && (((t - 1) >> 1) == ((s - 1) >> 1)));  // pair-local
}

__global__ __launch_bounds__(256)
void attn_kernel(const unsigned short* __restrict__ qb,
                 const unsigned short* __restrict__ kb,
                 const unsigned short* __restrict__ vb,
                 const float* __restrict__ bias,
                 unsigned short* __restrict__ ctxb) {
  extern __shared__ char smem[];
  float*          sc = (float*)smem;                        // 128x128 f32
  unsigned short* vT = (unsigned short*)(smem + 128 * 128 * 4);

  const int bid = blockIdx.x;
  const int n = bid >> 3;
  const int h = bid & 7;
  const int tid  = threadIdx.x;
  const int wid  = tid >> 5;
  const int lane = tid & 31;
  const int hf   = lane >> 4;
  const int r    = lane & 15;

  // Phase 1: scores = q.k^T + bias via WMMA, fragments direct from global.
  // 64 tiles (8x8 over 127x127 padded to 128), 8 per wave.
  for (int tt = wid; tt < 64; tt += 8) {
    const int s0 = (tt >> 3) << 4, t0 = (tt & 7) << 4;
    const unsigned short* qbase = &qb[(size_t)(n * S_ + s0 + r) * E_ + h * D_];
    const unsigned short* kbase = &kb[(size_t)(n * S_ + t0 + r) * E_ + h * D_];
    v8f acc = {};
#pragma unroll
    for (int kc = 0; kc < D_; kc += 32) {
      FragBF af, bf;
      af.u[0] = *(const uivec4*)&qbase[kc + hf * 8];
      af.u[1] = *(const uivec4*)&qbase[kc + 16 + hf * 8];
      bf.u[0] = *(const uivec4*)&kbase[kc + hf * 8];
      bf.u[1] = *(const uivec4*)&kbase[kc + 16 + hf * 8];
      acc = __builtin_amdgcn_wmma_f32_16x16x32_bf16(
          false, af.v, false, bf.v, (short)0, acc, false, false);
    }
#pragma unroll
    for (int i = 0; i < 8; ++i) {
      const int s = s0 + 8 * hf + i, t = t0 + r;
      if (s < S_ && t < S_)
        sc[s * 128 + t] = acc[i] + bias[((size_t)h * S_ + s) * S_ + t];
    }
  }

  // Stage v transposed into LDS (B-operand for phase 3); zero pad column.
  for (int p = tid; p < S_ * D_; p += 256) {
    const int t = p >> 7, d = p & 127;
    vT[d * VTP + t] = vb[(size_t)(n * S_ + t) * E_ + h * D_ + d];
  }
  for (int d = tid; d < 128; d += 256) vT[d * VTP + 127] = 0;  // kill pad NaNs
  __syncthreads();

  // Phase 2: w = 0.5*(softmax(sc) + softmax(mask ? -inf : sc)), wave per row
  for (int s = wid; s < S_; s += 8) {
    float gm = -1e30f, lm = -1e30f;
    for (int t = lane; t < S_; t += 32) {
      float v = sc[s * 128 + t];
      gm = fmaxf(gm, v);
      if (!is_masked(s, t)) lm = fmaxf(lm, v);
    }
    gm = wave_max(gm); lm = wave_max(lm);
    float gs = 0.f, ls = 0.f;
    for (int t = lane; t < S_; t += 32) {
      float v = sc[s * 128 + t];
      gs += __expf(v - gm);
      if (!is_masked(s, t)) ls += __expf(v - lm);
    }
    gs = wave_sum(gs); ls = wave_sum(ls);
    float ig = 1.f / gs, il = 1.f / ls;
    for (int t = lane; t < S_; t += 32) {
      float v = sc[s * 128 + t];
      sc[s * 128 + t] = 0.5f * (__expf(v - gm) * ig +
                                (is_masked(s, t) ? 0.f : __expf(v - lm) * il));
    }
    if (lane == 0) sc[s * 128 + 127] = 0.f;   // zero pad column
  }
  __syncthreads();

  // Phase 3: ctx = w @ v via WMMA. A-fragments packed bf16 from fp32 sc rows,
  // B-fragments contiguous from vT.
  for (int tt = wid; tt < 64; tt += 8) {
    const int s0 = (tt >> 3) << 4, d0 = (tt & 7) << 4;
    v8f acc = {};
#pragma unroll
    for (int kc = 0; kc < 128; kc += 32) {
      FragBF af;
      const float* wr = &sc[(s0 + r) * 128 + kc + hf * 8];
#pragma unroll
      for (int z = 0; z < 4; ++z) {
        af.u[0][z] = (unsigned)f2bf(wr[2 * z]) |
                     ((unsigned)f2bf(wr[2 * z + 1]) << 16);
        af.u[1][z] = (unsigned)f2bf(wr[16 + 2 * z]) |
                     ((unsigned)f2bf(wr[16 + 2 * z + 1]) << 16);
      }
      FragBF bf;
      const unsigned short* vr = &vT[(d0 + r) * VTP + kc + hf * 8];
      bf.u[0] = *(const uivec4*)vr;
      bf.u[1] = *(const uivec4*)(vr + 16);
      acc = __builtin_amdgcn_wmma_f32_16x16x32_bf16(
          false, af.v, false, bf.v, (short)0, acc, false, false);
    }
#pragma unroll
    for (int i = 0; i < 8; ++i) {
      const int s = s0 + 8 * hf + i, d = d0 + r;
      if (s < S_)
        ctxb[(size_t)(n * S_ + s) * E_ + h * D_ + d] = f2bf(acc[i]);
    }
  }
}

// ---------------------------------------------------------------------------
// x = LN(x + t) * g + b  (in place on x), also refresh bf16 mirror
// ---------------------------------------------------------------------------
__global__ __launch_bounds__(256)
void add_ln_kernel(float* __restrict__ x, const float* __restrict__ t,
                   const float* __restrict__ g, const float* __restrict__ b,
                   unsigned short* __restrict__ xb) {
  const int row = blockIdx.x, tid = threadIdx.x;
  const size_t base = (size_t)row * E_;
  float v[4];
  float s = 0.f, s2 = 0.f;
#pragma unroll
  for (int i = 0; i < 4; ++i) {
    int e = tid + i * 256;
    v[i] = x[base + e] + t[base + e];
    s += v[i]; s2 += v[i] * v[i];
  }
  __shared__ float red[16];
  s = wave_sum(s); s2 = wave_sum(s2);
  if ((tid & 31) == 0) { red[tid >> 5] = s; red[8 + (tid >> 5)] = s2; }
  __syncthreads();
  float ts = 0.f, ts2 = 0.f;
#pragma unroll
  for (int i = 0; i < 8; ++i) { ts += red[i]; ts2 += red[8 + i]; }
  float mean = ts * (1.0f / E_);
  float var  = ts2 * (1.0f / E_) - mean * mean;
  float inv  = rsqrtf(var + 1e-5f);
#pragma unroll
  for (int i = 0; i < 4; ++i) {
    int e = tid + i * 256;
    float y = (v[i] - mean) * inv * g[e] + b[e];
    x[base + e]  = y;
    xb[base + e] = f2bf(y);
  }
}

// ---------------------------------------------------------------------------
// out[n,:] = x[n, 0, :]
// ---------------------------------------------------------------------------
__global__ void out_copy_kernel(const float* __restrict__ x,
                                float* __restrict__ out) {
  int idx = blockIdx.x * 256 + threadIdx.x;
  if (idx >= NB_ * E_) return;
  int n = idx >> 10, e = idx & 1023;
  out[idx] = x[((size_t)n * S_) * E_ + e];
}

// ---------------------------------------------------------------------------
// Host launcher
// ---------------------------------------------------------------------------
extern "C" void kernel_launch(void* const* d_in, const int* in_sizes, int n_in,
                              void* d_out, int out_size, void* d_ws, size_t ws_size,
                              hipStream_t stream) {
  (void)in_sizes; (void)n_in; (void)out_size; (void)ws_size;

  const float* entity = (const float*)d_in[0];
  const float* cls    = (const float*)d_in[1];
  const float* ppe    = (const float*)d_in[2];
  const float* pln_g  = (const float*)d_in[3];
  const float* pln_b  = (const float*)d_in[4];
  const float* pqw    = (const float*)d_in[5];
  const float* pqb    = (const float*)d_in[6];
  const float* pkw    = (const float*)d_in[7];
  const float* pkb    = (const float*)d_in[8];
  auto LW = [&](int l, int j) { return (const float*)d_in[9 + l * 16 + j]; };
  // layer param order: q_w q_b k_w k_b v_w v_b o_w o_b ln1_g ln1_b fc1_w fc1_b fc2_w fc2_b ln2_g ln2_b

  // workspace bump allocator (256B aligned)
  char* ws = (char*)d_ws;
  size_t off = 0;
  auto alloc = [&](size_t bytes) -> void* {
    void* p = ws + off;
    off = (off + bytes + 255) & ~(size_t)255;
    return p;
  };
  const size_t MPAD = (size_t)M_ + 16;   // pad rows for attention edge tiles
  float*          x    = (float*)alloc((size_t)M_ * E_ * 4);
  float*          tmp  = (float*)alloc((size_t)M_ * E_ * 4);
  unsigned short* xb   = (unsigned short*)alloc((size_t)M_ * E_ * 2);
  unsigned short* qbuf = (unsigned short*)alloc(MPAD * E_ * 2);
  unsigned short* kbuf = (unsigned short*)alloc(MPAD * E_ * 2);
  unsigned short* vbuf = (unsigned short*)alloc(MPAD * E_ * 2);
  unsigned short* ctxb = (unsigned short*)alloc((size_t)M_ * E_ * 2);
  unsigned short* hbuf = (unsigned short*)alloc((size_t)M_ * FFN_ * 2);
  unsigned short* wq   = (unsigned short*)alloc((size_t)E_ * E_ * 2);
  unsigned short* wk   = (unsigned short*)alloc((size_t)E_ * E_ * 2);
  unsigned short* wv   = (unsigned short*)alloc((size_t)E_ * E_ * 2);
  unsigned short* wo   = (unsigned short*)alloc((size_t)E_ * E_ * 2);
  unsigned short* wf1  = (unsigned short*)alloc((size_t)FFN_ * E_ * 2);
  unsigned short* wf2  = (unsigned short*)alloc((size_t)E_ * FFN_ * 2);
  float*          biasT  = (float*)alloc((size_t)H_ * S_ * S_ * 4);
  float*          wgt128 = (float*)alloc((size_t)P_ * 128 * 4);
  float*          pqd    = (float*)alloc((size_t)H_ * P_ * 128 * 4);
  float*          pkd    = (float*)alloc((size_t)H_ * P_ * 128 * 4);

  auto conv = [&](const float* s, unsigned short* d, size_t n) {
    convert_bf16_kernel<<<(unsigned)CDIV(n, 256), 256, 0, stream>>>(s, d, n);
  };

  // ---- input assembly + positional bias precompute ----
  init_x_kernel<<<(unsigned)CDIV((size_t)M_ * E_, 256), 256, 0, stream>>>(entity, cls, x, xb);
  ln_pos_kernel<<<P_, 128, 0, stream>>>(ppe, pln_g, pln_b, wgt128);
  pos_lin_kernel<<<dim3(P_, H_), 128, 0, stream>>>(wgt128, pqw, pqb, pqd, QK_SCALE);
  pos_lin_kernel<<<dim3(P_, H_), 128, 0, stream>>>(wgt128, pkw, pkb, pkd, 1.0f);
  bias_build_kernel<<<dim3(H_, S_), 128, 0, stream>>>(pqd, pkd, biasT);

  const dim3 gemmBlk(256);
  const dim3 gE(E_ / BN, M_ / BM);      // (8, 254)
  const dim3 gF(FFN_ / BN, M_ / BM);    // (32, 254)
  const unsigned ropeGrid = (unsigned)CDIV((size_t)M_ * E_ / 2, 256);

  for (int l = 0; l < 3; ++l) {
    // stage weights as bf16
    conv(LW(l, 0),  wq,  (size_t)E_ * E_);
    conv(LW(l, 2),  wk,  (size_t)E_ * E_);
    conv(LW(l, 4),  wv,  (size_t)E_ * E_);
    conv(LW(l, 6),  wo,  (size_t)E_ * E_);
    conv(LW(l, 10), wf1, (size_t)FFN_ * E_);
    conv(LW(l, 12), wf2, (size_t)E_ * FFN_);

    // QKV projections (WMMA), bf16 out
    gemm_bf16_kernel<false, true><<<gE, gemmBlk, 0, stream>>>(xb, wq, LW(l, 1), nullptr, qbuf, M_, E_, E_);
    gemm_bf16_kernel<false, true><<<gE, gemmBlk, 0, stream>>>(xb, wk, LW(l, 3), nullptr, kbuf, M_, E_, E_);
    gemm_bf16_kernel<false, true><<<gE, gemmBlk, 0, stream>>>(xb, wv, LW(l, 5), nullptr, vbuf, M_, E_, E_);

    // RoPE (+q scaling)
    rope_kernel<<<ropeGrid, 256, 0, stream>>>(qbuf, QK_SCALE);
    rope_kernel<<<ropeGrid, 256, 0, stream>>>(kbuf, 1.0f);
    rope_kernel<<<ropeGrid, 256, 0, stream>>>(vbuf, 1.0f);

    // Attention (WMMA scores + dual softmax + WMMA ctx)
    attn_kernel<<<NB_ * H_, 256, ATTN_LDS, stream>>>(qbuf, kbuf, vbuf, biasT, ctxb);

    // Output projection (WMMA) -> tmp f32; residual + LN1
    gemm_bf16_kernel<false, false><<<gE, gemmBlk, 0, stream>>>(ctxb, wo, LW(l, 7), tmp, nullptr, M_, E_, E_);
    add_ln_kernel<<<M_, 256, 0, stream>>>(x, tmp, LW(l, 8), LW(l, 9), xb);

    // FFN: fc1 (WMMA, fused ReLU, bf16 out) ; fc2 (WMMA, f32 out) ; LN2
    gemm_bf16_kernel<true, true><<<gF, gemmBlk, 0, stream>>>(xb, wf1, LW(l, 11), nullptr, hbuf, M_, FFN_, E_);
    gemm_bf16_kernel<false, false><<<gE, gemmBlk, 0, stream>>>(hbuf, wf2, LW(l, 13), tmp, nullptr, M_, E_, FFN_);
    add_ln_kernel<<<M_, 256, 0, stream>>>(x, tmp, LW(l, 14), LW(l, 15), xb);
  }

  out_copy_kernel<<<CDIV(NB_ * E_, 256), 256, 0, stream>>>(x, (float*)d_out);
}
